// GraphAttentionLayer_41077067219209
// MI455X (gfx1250) — compile-verified, matrix operand order checked by reading, not run on hardware
//
#include <hip/hip_runtime.h>
#include <hip/hip_bf16.h>
#include <cstdint>

#define B_     8
#define S_     2048
#define DIN_   256
#define DOUT_  256
#define ALPHA_ 0.2f
#define NEG_   -9.0e15f

typedef __attribute__((ext_vector_type(16))) _Float16 v16h;
typedef __attribute__((ext_vector_type(8)))  float    v8f;

// ---------------------------------------------------------------------------
// Kernel 1: h = x @ W^T via v_wmma_f32_16x16x32_f16 (f32 accumulate).
// Emits hT (f16, [B][DOUT][S]) so kernel 2's B-operand loads are contiguous,
// and f_i = h@a1, f_j = h@a2 (f32) via an LDS reduction of the h tile.
// Grid: B*(S/16) blocks of 256 threads (8 waves); wave w owns N-tiles 2w,2w+1.
// ---------------------------------------------------------------------------
__global__ __launch_bounds__(256)
void gat_proj_kernel(const float* __restrict__ x,
                     const float* __restrict__ W,
                     const float* __restrict__ a,
                     _Float16* __restrict__ hT,
                     float* __restrict__ f_i,
                     float* __restrict__ f_j)
{
    __shared__ float s_h[16][DOUT_];   // 16KB f32 h tile (for f_i/f_j)
    __shared__ float s_rfi[16][16];
    __shared__ float s_rfj[16][16];

    const int tid  = threadIdx.x;
    const int lane = tid & 31;
    const int wv   = tid >> 5;                 // 0..7
    const int blk  = blockIdx.x;
    const int b    = blk / (S_ / 16);
    const int i0   = (blk % (S_ / 16)) * 16;
    const int rowA = lane & 15;                // A-row / B-column selector
    const int hi   = lane >> 4;                // K-half selector (16-bit layout)

    const float* xrow = x + ((size_t)b * S_ + (size_t)(i0 + rowA)) * DIN_;

    for (int t = 0; t < 2; ++t) {
        const int n0 = (wv * 2 + t) * 16;
        const float* wrow = W + (size_t)(n0 + rowA) * DIN_;  // B col n = W row n
        v8f acc = {};
        for (int kt = 0; kt < DIN_ / 32; ++kt) {
            const int kb = kt * 32 + hi * 8;
            v16h af, bf;
#pragma unroll
            for (int e = 0; e < 8; ++e) {
                // elems 0..7 -> K = kb+e ; elems 8..15 -> K = kb+16+e
                af[e]     = (_Float16)xrow[kb + e];
                af[e + 8] = (_Float16)xrow[kb + 16 + e];
                bf[e]     = (_Float16)wrow[kb + e];
                bf[e + 8] = (_Float16)wrow[kb + 16 + e];
            }
            acc = __builtin_amdgcn_wmma_f32_16x16x32_f16(
                      false, af, false, bf, (short)0, acc, false, false);
        }
        // C/D layout: lane (hi,n=rowA), VGPR v -> M = v + 8*hi
        const int n = n0 + rowA;
        _Float16* hTrow = hT + ((size_t)b * DOUT_ + n) * S_ + i0 + hi * 8;
#pragma unroll
        for (int v = 0; v < 8; ++v) {
            const int m = v + hi * 8;
            s_h[m][n] = acc[v];
            hTrow[v]  = (_Float16)acc[v];   // 8 contiguous halfs
        }
    }
    __syncthreads();

    // f_i / f_j: 256 threads -> (row rr, 16-wide chunk cc) partials, then fold
    const int rr = tid >> 4, cc = tid & 15;
    float pfi = 0.f, pfj = 0.f;
#pragma unroll
    for (int o = 0; o < 16; ++o) {
        const int oo = cc * 16 + o;
        const float hv = s_h[rr][oo];
        pfi += hv * a[oo];
        pfj += hv * a[DOUT_ + oo];
    }
    s_rfi[rr][cc] = pfi; s_rfj[rr][cc] = pfj;
    __syncthreads();
    if (tid < 16) {
        float fi = 0.f, fj = 0.f;
#pragma unroll
        for (int c = 0; c < 16; ++c) { fi += s_rfi[tid][c]; fj += s_rfj[tid][c]; }
        f_i[(size_t)b * S_ + i0 + tid] = fi;
        f_j[(size_t)b * S_ + i0 + tid] = fj;
    }
}

// ---------------------------------------------------------------------------
// Kernel 2: fused masked-softmax attention, flash style (single pass over adj,
// never materializes the 134MB attention matrix). Per block: 16 output rows,
// 4 waves x 4 N-tiles = all 256 columns. Online max/sum; P@H accumulated with
// v_wmma_f32_16x16x32_f16 (K = 32 j-columns per iteration).
// ---------------------------------------------------------------------------
__global__ __launch_bounds__(128)
void gat_attn_kernel(const int* __restrict__ adj,
                     const _Float16* __restrict__ hT,
                     const float* __restrict__ f_i,
                     const float* __restrict__ f_j,
                     float* __restrict__ out)
{
    __shared__ float    s_sc[16][32];
    __shared__ _Float16 s_p[16][32];
    __shared__ float    s_fj[32];
    __shared__ float    s_fi[16];
    __shared__ float    s_M[16];
    __shared__ float    s_L[16];
    __shared__ float    s_c[16];

    const int tid  = threadIdx.x;
    const int lane = tid & 31;
    const int wv   = tid >> 5;                 // 0..3
    const int blk  = blockIdx.x;
    const int b    = blk / (S_ / 16);
    const int i0   = (blk % (S_ / 16)) * 16;
    const int rowA = lane & 15;
    const int hi   = lane >> 4;
    const size_t bS = (size_t)b * S_;

    if (tid < 16) {
        s_fi[tid] = f_i[bS + i0 + tid];
        s_M[tid]  = -INFINITY;
        s_L[tid]  = 0.f;
    }

    v8f acc[4] = {};
    for (int jt = 0; jt < S_ / 32; ++jt) {
        const int j0 = jt * 32;
        if (tid < 32) s_fj[tid] = f_j[bS + j0 + tid];
        __syncthreads();

        // --- scores: e = leakyrelu(fi+fj), masked by adj (16x32 tile) ---
#pragma unroll
        for (int q = 0; q < 4; ++q) {
            const int e  = tid + q * 128;
            const int ii = e >> 5;
            const int jj = e & 31;
            const int* arow = adj + (bS + i0 + ii) * S_ + j0;
            float sc = s_fi[ii] + s_fj[jj];
            sc = (sc >= 0.f) ? sc : (ALPHA_ * sc);
            sc = (arow[jj] > 0) ? sc : NEG_;
            s_sc[ii][jj] = sc;
            if (j0 + 32 < S_) __builtin_prefetch(arow + 32 + jj, 0, 0);
        }
        __syncthreads();

        // --- online softmax update, one lane per row ---
        if (tid < 16) {
            const float Mo = s_M[tid];
            float mt = -INFINITY;
#pragma unroll
            for (int jj = 0; jj < 32; ++jj) mt = fmaxf(mt, s_sc[tid][jj]);
            const float Mn = fmaxf(Mo, mt);
            const float c  = __expf(Mo - Mn);     // first iter: exp(-inf)=0
            float sum = 0.f;
#pragma unroll
            for (int jj = 0; jj < 32; ++jj) {
                const float p = __expf(s_sc[tid][jj] - Mn);
                s_p[tid][jj] = (_Float16)p;
                sum += p;
            }
            s_M[tid] = Mn;
            s_L[tid] = s_L[tid] * c + sum;
            s_c[tid] = c;
        }
        __syncthreads();

        // --- rescale accumulators and accumulate P(16x32) @ H(32x16) ---
        float cs[8];
#pragma unroll
        for (int v = 0; v < 8; ++v) cs[v] = s_c[v + hi * 8];
        v16h af;
#pragma unroll
        for (int e = 0; e < 8; ++e) {
            af[e]     = s_p[rowA][hi * 8 + e];
            af[e + 8] = s_p[rowA][16 + hi * 8 + e];
        }
#pragma unroll
        for (int t = 0; t < 4; ++t) {
            const int n = (wv * 4 + t) * 16 + rowA;
            const _Float16* hrow = hT + ((size_t)b * DOUT_ + n) * S_ + j0 + hi * 8;
            v16h bf;
#pragma unroll
            for (int e = 0; e < 8; ++e) {
                bf[e]     = hrow[e];        // K = j0 + hi*8 + e
                bf[e + 8] = hrow[16 + e];   // K = j0 + 16 + hi*8 + e
            }
#pragma unroll
            for (int v = 0; v < 8; ++v) acc[t][v] *= cs[v];
            acc[t] = __builtin_amdgcn_wmma_f32_16x16x32_f16(
                         false, af, false, bf, (short)0, acc[t], false, false);
        }
        __syncthreads();
    }

    // --- normalize by row sums and store f32 output ---
    float invL[8];
#pragma unroll
    for (int v = 0; v < 8; ++v) invL[v] = 1.f / s_L[v + hi * 8];
#pragma unroll
    for (int t = 0; t < 4; ++t) {
        const int n = (wv * 4 + t) * 16 + rowA;
#pragma unroll
        for (int v = 0; v < 8; ++v) {
            const int m = v + hi * 8;
            out[(bS + i0 + m) * DOUT_ + n] = acc[t][v] * invL[v];
        }
    }
}

// ---------------------------------------------------------------------------
extern "C" void kernel_launch(void* const* d_in, const int* in_sizes, int n_in,
                              void* d_out, int out_size, void* d_ws, size_t ws_size,
                              hipStream_t stream) {
    const float* x   = (const float*)d_in[0];   // (B,S,DIN)  f32
    const int*   adj = (const int*)d_in[1];     // (B,S,S)    i32
    const float* W   = (const float*)d_in[2];   // (DOUT,DIN) f32
    const float* a   = (const float*)d_in[3];   // (1,2*DOUT) f32
    float* out = (float*)d_out;                 // (B,S,DOUT) f32

    // workspace: hT f16 (8MB) | f_i f32 (64KB) | f_j f32 (64KB)
    char* ws = (char*)d_ws;
    _Float16* hT = (_Float16*)ws;
    float* f_i = (float*)(ws + (size_t)B_ * DOUT_ * S_ * sizeof(_Float16));
    float* f_j = f_i + (size_t)B_ * S_;

    const int nblocks = B_ * (S_ / 16);   // 1024
    gat_proj_kernel<<<nblocks, 256, 0, stream>>>(x, W, a, hT, f_i, f_j);
    gat_attn_kernel<<<nblocks, 128, 0, stream>>>(adj, hT, f_i, f_j, out);
}
